// CrossViewAttention3D_36043365548298
// MI455X (gfx1250) — matrix operand order repaired and checked by reading, not compiled, not run on hardware
//
#include <hip/hip_runtime.h>
#include <hip/hip_bf16.h>

typedef __attribute__((ext_vector_type(16))) __bf16 v16bf;
typedef __attribute__((ext_vector_type(8)))  __bf16 v8bf;
typedef __attribute__((ext_vector_type(8)))  float  v8f;

#define C_DIM 512
#define N_SEQ 4096
#define HEADS 8
#define HDIM  64
#define QWAVES 8                 // waves per attention workgroup
#define NCH (N_SEQ / 32)         // 32-key chunks

// ---------------------------------------------------------------------------
// WMMA helper: D = A(16x32 bf16) x B(32x16 bf16) + C(16x16 f32)
// ---------------------------------------------------------------------------
__device__ __forceinline__ v8f wmma_bf16(v16bf a, v16bf b, v8f c) {
  return __builtin_amdgcn_wmma_f32_16x16x32_bf16(false, a, false, b, (short)0, c,
                                                 false, false);
}

// A-matrix 16x32 bf16 load from row-major src (ld = elements per row).
// Contiguous k order used here (not ISA table order): lane m = lane&15; half h = lane>>4 holds K = 16h + 0..15
// -> two contiguous 16-byte loads per lane.
__device__ __forceinline__ v16bf load_a(const __bf16* src, int ld, int row0,
                                        int k0, int lane) {
  const int lm = lane & 15, h = lane >> 4;
  const __bf16* p = src + (size_t)(row0 + lm) * ld + k0 + 16 * h;
  union { v16bf v; v8bf x[2]; } u;
  u.x[0] = *(const v8bf*)p;
  u.x[1] = *(const v8bf*)(p + 8);
  return u.v;
}

// B-matrix 32x16 bf16 load (global). Element (k,n) at colbase + n*ldcol + k.
// Lane n = lane&15 holds K = 16*half + 0..15 -> one contiguous 32-byte run.
__device__ __forceinline__ v16bf load_b(const __bf16* colbase, size_t ldcol,
                                        int lane) {
  const int lm = lane & 15, h = lane >> 4;
  return *(const v16bf*)(colbase + (size_t)lm * ldcol + 16 * h);
}

// Same B-tile fetch out of LDS (two ds_load_b128 per lane).
__device__ __forceinline__ v16bf load_b_lds(const __bf16* colbase, int ldcol,
                                            int lane) {
  const int lm = lane & 15, h = lane >> 4;
  const __bf16* p = colbase + lm * ldcol + 16 * h;
  union { v16bf v; v8bf x[2]; } u;
  u.x[0] = *(const v8bf*)p;
  u.x[1] = *(const v8bf*)(p + 8);
  return u.v;
}

// ---------------------------------------------------------------------------
// Stage 0a: convert the four 512x512 weight matrices fp32 -> bf16 (row-major)
// ---------------------------------------------------------------------------
__global__ void cva_convert_weights(const float* __restrict__ Wq,
                                    const float* __restrict__ Wk,
                                    const float* __restrict__ Wv,
                                    const float* __restrict__ Wo,
                                    __bf16* __restrict__ q, __bf16* __restrict__ k,
                                    __bf16* __restrict__ v, __bf16* __restrict__ o) {
  const int idx = blockIdx.x * blockDim.x + threadIdx.x;  // 0 .. 512*512-1
  q[idx] = (__bf16)Wq[idx];
  k[idx] = (__bf16)Wk[idx];
  v[idx] = (__bf16)Wv[idx];
  o[idx] = (__bf16)Wo[idx];
}

// ---------------------------------------------------------------------------
// Stage 0b: transpose inputs [C][N] fp32 -> [N][C] bf16 (both feature maps)
// ---------------------------------------------------------------------------
__global__ void cva_transpose_inputs(const float* __restrict__ Xv,
                                     const float* __restrict__ Xr,
                                     __bf16* __restrict__ Xvt,
                                     __bf16* __restrict__ Xrt) {
  const int idx = blockIdx.x * blockDim.x + threadIdx.x;  // n*512 + c
  const float* X = blockIdx.y ? Xr : Xv;
  __bf16* Xt = blockIdx.y ? Xrt : Xvt;
  const int c = idx & (C_DIM - 1);
  const int n = idx >> 9;
  Xt[idx] = (__bf16)X[(size_t)c * N_SEQ + n];
}

// ---------------------------------------------------------------------------
// Stage 1: fused QKV projection. One wave computes one 16(cout) x 16(n) tile.
// blockIdx.z: 0=Q (from video), 1=K (from ref), 2=V (from ref).
// Q,K stored [N][C] bf16; V stored transposed [C][N] bf16.
// ---------------------------------------------------------------------------
__global__ __launch_bounds__(32) void cva_proj_qkv(
    const __bf16* __restrict__ Wqb, const __bf16* __restrict__ Wkb,
    const __bf16* __restrict__ Wvb, const float* __restrict__ bq,
    const float* __restrict__ bk, const float* __restrict__ bv,
    const __bf16* __restrict__ Xvt, const __bf16* __restrict__ Xrt,
    __bf16* __restrict__ Qb, __bf16* __restrict__ Kb, __bf16* __restrict__ Vt) {
  const int ct = blockIdx.x;       // 0..31  (output-channel tile)
  const int nt = blockIdx.y;       // 0..255 (token tile)
  const int which = blockIdx.z;    // 0=Q,1=K,2=V
  const int lane = threadIdx.x, lm = lane & 15, h = lane >> 4;

  const __bf16* W = (which == 0) ? Wqb : (which == 1) ? Wkb : Wvb;
  const float* bias = (which == 0) ? bq : (which == 1) ? bk : bv;
  const __bf16* X = (which == 0) ? Xvt : Xrt;

  v8f acc = {};
#pragma unroll
  for (int ks = 0; ks < C_DIM / 32; ++ks) {
    v16bf a = load_a(W, C_DIM, ct * 16, ks * 32, lane);
    v16bf b = load_b(X + (size_t)nt * 16 * C_DIM + ks * 32, C_DIM, lane);
    acc = wmma_bf16(a, b, acc);
  }

  if (which < 2) {
    __bf16* dst = (which == 0) ? Qb : Kb;
    v8bf pk;
#pragma unroll
    for (int r = 0; r < 8; ++r)
      pk[r] = (__bf16)(acc[r] + bias[ct * 16 + 8 * h + r]);
    *(v8bf*)(dst + (size_t)(nt * 16 + lm) * C_DIM + ct * 16 + 8 * h) = pk;
  } else {
#pragma unroll
    for (int r = 0; r < 8; ++r) {
      const int cg = ct * 16 + 8 * h + r;
      Vt[(size_t)cg * N_SEQ + nt * 16 + lm] = (__bf16)(acc[r] + bias[cg]);
    }
  }
}

// ---------------------------------------------------------------------------
// Stage 2: flash attention, 8 waves / workgroup. K/V chunks are staged into
// LDS with CDNA5 async-to-LDS DMA (ASYNCcnt), double-buffered so the next
// chunk's DMA overlaps this chunk's WMMAs + softmax.
//   kbuf: [key][dim] (dim contiguous)  -> B tiles for Q*K^T
//   vbuf: [dim][key] (key contiguous)  -> B tiles for P*V
// ---------------------------------------------------------------------------
__device__ __forceinline__ void cva_issue_chunk(const __bf16* __restrict__ Kb,
                                                const __bf16* __restrict__ Vt,
                                                int head, int key0,
                                                __bf16* kbuf, __bf16* vbuf,
                                                int tid) {
  // K: 32 keys x 64 dims = 256 b128 transfers; one per thread
  {
    const int key = tid >> 3, seg = tid & 7;
    const __bf16* g = Kb + (size_t)(key0 + key) * C_DIM + head * HDIM + seg * 8;
    const unsigned l = (unsigned)(size_t)(kbuf + key * HDIM + seg * 8);
    asm volatile("global_load_async_to_lds_b128 %0, %1, off"
                 :: "v"(l), "v"(g) : "memory");
  }
  // V: 64 dims x 32 keys = 256 b128 transfers; one per thread
  {
    const int dim = tid >> 2, seg = tid & 3;
    const __bf16* g = Vt + (size_t)(head * HDIM + dim) * N_SEQ + key0 + seg * 8;
    const unsigned l = (unsigned)(size_t)(vbuf + dim * 32 + seg * 8);
    asm volatile("global_load_async_to_lds_b128 %0, %1, off"
                 :: "v"(l), "v"(g) : "memory");
  }
}

__global__ __launch_bounds__(32 * QWAVES) void cva_attention(
    const __bf16* __restrict__ Qb, const __bf16* __restrict__ Kb,
    const __bf16* __restrict__ Vt, __bf16* __restrict__ Ob) {
  const int head = blockIdx.y;
  const int tid = threadIdx.x;
  const int wave = tid >> 5;
  const int lane = tid & 31;
  const int lm = lane & 15, h = lane >> 4;

  __shared__ __attribute__((aligned(16))) __bf16 kbuf[2][32 * HDIM];
  __shared__ __attribute__((aligned(16))) __bf16 vbuf[2][HDIM * 32];
  __shared__ __attribute__((aligned(16))) __bf16 plds[QWAVES][16 * 32];

  const int q0 = (blockIdx.x * QWAVES + wave) * 16;
  const v16bf qa0 = load_a(Qb, C_DIM, q0, head * HDIM + 0, lane);
  const v16bf qa1 = load_a(Qb, C_DIM, q0, head * HDIM + 32, lane);

  float mrun[8], lrun[8];
  v8f o0 = {}, o1 = {}, o2 = {}, o3 = {};
#pragma unroll
  for (int r = 0; r < 8; ++r) { mrun[r] = -__builtin_inff(); lrun[r] = 0.f; }

  cva_issue_chunk(Kb, Vt, head, 0, kbuf[0], vbuf[0], tid);

  const float scale = 0.125f;  // HEAD_DIM^-0.5
  for (int ch = 0; ch < NCH; ++ch) {
    const int b = ch & 1;
    asm volatile("s_wait_asynccnt 0x0" ::: "memory");  // my DMA for buf[b] done
    __syncthreads();                                   // publish to all waves
    if (ch + 1 < NCH)
      cva_issue_chunk(Kb, Vt, head, (ch + 1) * 32, kbuf[1 - b], vbuf[1 - b], tid);

    const __bf16* kb = kbuf[b];
    v8f s0 = {}, s1 = {};
    s0 = wmma_bf16(qa0, load_b_lds(kb, HDIM, lane), s0);
    s0 = wmma_bf16(qa1, load_b_lds(kb + 32, HDIM, lane), s0);
    s1 = wmma_bf16(qa0, load_b_lds(kb + 16 * HDIM, HDIM, lane), s1);
    s1 = wmma_bf16(qa1, load_b_lds(kb + 16 * HDIM + 32, HDIM, lane), s1);

    __bf16* pl = plds[wave];
#pragma unroll
    for (int r = 0; r < 8; ++r) {
      const float a0 = s0[r] * scale;
      const float a1 = s1[r] * scale;
      // row m = r + 8*h lives in this 16-lane half: butterfly over 16 lanes
      float mx = fmaxf(a0, a1);
      mx = fmaxf(mx, __shfl_xor(mx, 1, 32));
      mx = fmaxf(mx, __shfl_xor(mx, 2, 32));
      mx = fmaxf(mx, __shfl_xor(mx, 4, 32));
      mx = fmaxf(mx, __shfl_xor(mx, 8, 32));
      const float mnew = fmaxf(mrun[r], mx);
      const float alpha = __expf(mrun[r] - mnew);
      const float p0 = __expf(a0 - mnew);
      const float p1 = __expf(a1 - mnew);
      float ls = p0 + p1;
      ls += __shfl_xor(ls, 1, 32);
      ls += __shfl_xor(ls, 2, 32);
      ls += __shfl_xor(ls, 4, 32);
      ls += __shfl_xor(ls, 8, 32);
      lrun[r] = lrun[r] * alpha + ls;
      mrun[r] = mnew;
      o0[r] *= alpha; o1[r] *= alpha; o2[r] *= alpha; o3[r] *= alpha;
      const int m = r + 8 * h;
      pl[m * 32 + lm] = (__bf16)p0;
      pl[m * 32 + 16 + lm] = (__bf16)p1;
    }
    __syncthreads();  // plds visible; also fences buf[b] reads vs next DMA
    union { v16bf v; v8bf x[2]; } pu;
    pu.x[0] = *(const v8bf*)&pl[lm * 32 + 16 * h];
    pu.x[1] = *(const v8bf*)&pl[lm * 32 + 16 * h + 8];
    const v16bf pa = pu.v;

    const __bf16* vb = vbuf[b];
    o0 = wmma_bf16(pa, load_b_lds(vb + 0 * 16 * 32, 32, lane), o0);
    o1 = wmma_bf16(pa, load_b_lds(vb + 1 * 16 * 32, 32, lane), o1);
    o2 = wmma_bf16(pa, load_b_lds(vb + 2 * 16 * 32, 32, lane), o2);
    o3 = wmma_bf16(pa, load_b_lds(vb + 3 * 16 * 32, 32, lane), o3);
  }

#pragma unroll
  for (int r = 0; r < 8; ++r) {
    const float inv = 1.0f / lrun[r];
    const int m = r + 8 * h;
    __bf16* dst = Ob + (size_t)(q0 + m) * C_DIM + head * HDIM + lm;
    dst[0]  = (__bf16)(o0[r] * inv);
    dst[16] = (__bf16)(o1[r] * inv);
    dst[32] = (__bf16)(o2[r] * inv);
    dst[48] = (__bf16)(o3[r] * inv);
  }
}

// ---------------------------------------------------------------------------
// Stage 3: output projection Y[C][N] = Wo @ O^T + bo, fp32 out (=B,C,T,H,W)
// ---------------------------------------------------------------------------
__global__ __launch_bounds__(32) void cva_out_proj(
    const __bf16* __restrict__ Wob, const float* __restrict__ bo,
    const __bf16* __restrict__ Ob, float* __restrict__ out) {
  const int ct = blockIdx.x;  // 0..31
  const int nt = blockIdx.y;  // 0..255
  const int lane = threadIdx.x, lm = lane & 15, h = lane >> 4;

  v8f acc = {};
#pragma unroll
  for (int ks = 0; ks < C_DIM / 32; ++ks) {
    v16bf a = load_a(Wob, C_DIM, ct * 16, ks * 32, lane);
    v16bf b = load_b(Ob + (size_t)nt * 16 * C_DIM + ks * 32, C_DIM, lane);
    acc = wmma_bf16(a, b, acc);
  }
#pragma unroll
  for (int r = 0; r < 8; ++r) {
    const int cg = ct * 16 + 8 * h + r;
    out[(size_t)cg * N_SEQ + nt * 16 + lm] = acc[r] + bo[cg];
  }
}

// ---------------------------------------------------------------------------
extern "C" void kernel_launch(void* const* d_in, const int* in_sizes, int n_in,
                              void* d_out, int out_size, void* d_ws, size_t ws_size,
                              hipStream_t stream) {
  const float* video = (const float*)d_in[0];
  const float* ref   = (const float*)d_in[1];
  const float* Wq = (const float*)d_in[2];
  const float* bq = (const float*)d_in[3];
  const float* Wk = (const float*)d_in[4];
  const float* bk = (const float*)d_in[5];
  const float* Wv = (const float*)d_in[6];
  const float* bv = (const float*)d_in[7];
  const float* Wo = (const float*)d_in[8];
  const float* bo = (const float*)d_in[9];
  float* out = (float*)d_out;

  char* ws = (char*)d_ws;
  size_t off = 0;
  auto alloc = [&](size_t bytes) -> char* {
    char* p = ws + off;
    off += (bytes + 255) & ~(size_t)255;
    return p;
  };
  const size_t xcnt = (size_t)N_SEQ * C_DIM;  // 2M elements
  __bf16* Xvt = (__bf16*)alloc(xcnt * 2);
  __bf16* Xrt = (__bf16*)alloc(xcnt * 2);
  __bf16* Wqb = (__bf16*)alloc((size_t)C_DIM * C_DIM * 2);
  __bf16* Wkb = (__bf16*)alloc((size_t)C_DIM * C_DIM * 2);
  __bf16* Wvb = (__bf16*)alloc((size_t)C_DIM * C_DIM * 2);
  __bf16* Wob = (__bf16*)alloc((size_t)C_DIM * C_DIM * 2);
  __bf16* Qb  = (__bf16*)alloc(xcnt * 2);   // [N][C]
  __bf16* Kb  = (__bf16*)alloc(xcnt * 2);   // [N][C]
  __bf16* Vt  = (__bf16*)alloc(xcnt * 2);   // [C][N]
  __bf16* Ob  = (__bf16*)alloc(xcnt * 2);   // [N][C]
  (void)ws_size; (void)in_sizes; (void)n_in; (void)out_size;

  cva_convert_weights<<<dim3((C_DIM * C_DIM) / 256), dim3(256), 0, stream>>>(
      Wq, Wk, Wv, Wo, Wqb, Wkb, Wvb, Wob);
  cva_transpose_inputs<<<dim3(xcnt / 256, 2), dim3(256), 0, stream>>>(
      video, ref, Xvt, Xrt);
  cva_proj_qkv<<<dim3(C_DIM / 16, N_SEQ / 16, 3), dim3(32), 0, stream>>>(
      Wqb, Wkb, Wvb, bq, bk, bv, Xvt, Xrt, Qb, Kb, Vt);
  cva_attention<<<dim3(N_SEQ / (16 * QWAVES), HEADS), dim3(32 * QWAVES), 0, stream>>>(
      Qb, Kb, Vt, Ob);
  cva_out_proj<<<dim3(C_DIM / 16, N_SEQ / 16), dim3(32), 0, stream>>>(
      Wob, bo, Ob, out);
}
